// BatchDropTop_1211180778377
// MI455X (gfx1250) — compile-verified, matrix operand order checked by reading, not run on hardware
//
#include <hip/hip_runtime.h>

// ---------------------------------------------------------------------------
// BatchDropTop for x[64, 2048, 24, 8] f32 on MI455X (gfx1250, wave32).
//
// Key insight: the L2-normalization in the reference is a positive per-batch
// scalar -> it cannot change which 8 rows have the largest max activation.
// So the computation reduces to:
//   1) rowstat[b,h] = max_w sum_c x[b,c,h,w]^2      (100 MB read)
//   2) top-8 rows per batch -> mask[b,h] in {0,1}   (tiny)
//   3) out = x * mask[b,h]                          (100 MB read + 100 MB write)
// x fits in the 192 MB L2, so pass 3's read mostly hits L2; NT stores keep
// the output from evicting x. HBM floor ~200 MB ~ 8.6 us @ 23.3 TB/s.
//
// The channel reduction is mapped onto v_wmma_f32_16x16x32_f16 with a
// ones A-matrix: D[m,n] = sum_k B[k,n], i.e. 16 column sums per instruction.
// Column sums are invariant to the K-slot permutation of the B layout
// (columns are indexed by lane&15), so this is robust to layout details,
// and every C component holds the same sum -> acc[0], no diag extraction.
// ---------------------------------------------------------------------------

typedef __attribute__((ext_vector_type(16))) _Float16 v16h;
typedef __attribute__((ext_vector_type(8)))  float    v8f;
typedef __attribute__((ext_vector_type(4)))  float    v4f;

#define B_DIM  64
#define C_DIM  2048
#define H_DIM  24
#define W_DIM  8
#define HW     192            // h*w spatial sites per (b,c) plane
#define RH     8              // round(0.33 * 24)
#define NCHUNK 8              // channel chunks (parallelism, no atomics)
#define CCHUNK (C_DIM / NCHUNK)  // 256 channels per chunk
#define NTILE  (HW / 16)      // 12 position tiles of 16 sites

// Kernel 1: partial sum-of-squares over channels via WMMA column sums.
// grid = (64 batches, 12 position tiles, 8 channel chunks), block = 1 wave.
__global__ __launch_bounds__(32)
void bdt_stats_wmma(const float* __restrict__ x, float* __restrict__ psum) {
  const int b    = blockIdx.x;
  const int tile = blockIdx.y;
  const int cblk = blockIdx.z;
  const int lane = threadIdx.x;        // 0..31, full wave, no divergence
  const int n    = lane & 15;          // B-matrix column == position in tile
  const int khi  = (lane >> 4) * 16;   // this lane holds 16 of the 32 K values
  const int p    = tile * 16 + n;      // spatial site 0..191

  const float* xp = x + (size_t)b * C_DIM * HW + p;   // stride HW per channel

  v16h ones;
#pragma unroll
  for (int i = 0; i < 16; ++i) ones[i] = (_Float16)1.0f;

  v8f acc = {};  // f32 accumulator across all WMMAs of this chunk

  int c0 = cblk * CCHUNK;
  const int c1 = c0 + CCHUNK;
  for (; c0 < c1; c0 += 32) {          // 32 channels per WMMA
    v16h yb;
#pragma unroll
    for (int i = 0; i < 16; ++i) {
      // coalesced: fixed i -> 16 lanes read 64B contiguous
      float v = xp[(size_t)(c0 + khi + i) * HW];
      yb[i] = (_Float16)(v * v);       // ranking statistic only: f16 is fine
    }
    // D[m,n] += sum_k 1 * yb[k,n]  -> column sums, accumulated in f32
    acc = __builtin_amdgcn_wmma_f32_16x16x32_f16(false, ones, false, yb,
                                                 (short)0, acc, false, false);
  }

  // Every C component equals the column sum for position p; lanes 0..15 cover
  // the 16 distinct columns. One plain store per (b, cblk, p): deterministic.
  if (lane < 16) {
    psum[((size_t)b * NCHUNK + cblk) * HW + p] = acc[0];
  }
}

// Kernel 2: per-row max over w (after summing the 8 channel chunks in fixed
// order), then rank-based top-8 selection. grid = 64 blocks, block = 1 wave.
__global__ __launch_bounds__(32)
void bdt_topk_mask(const float* __restrict__ psum, float* __restrict__ mask) {
  const int b = blockIdx.x;
  const int l = threadIdx.x;
  __shared__ float rmax[H_DIM];
  if (l < H_DIM) {
    float mx = -3.402823466e38f;
    for (int w = 0; w < W_DIM; ++w) {
      float s = 0.0f;
      for (int cb = 0; cb < NCHUNK; ++cb)
        s += psum[((size_t)b * NCHUNK + cb) * HW + l * W_DIM + w];
      mx = fmaxf(mx, s);
    }
    rmax[l] = mx;
  }
  __syncthreads();
  if (l < H_DIM) {
    const float v = rmax[l];
    int rank = 0;   // number of rows strictly ahead of me (argsort tie-break)
    for (int j = 0; j < H_DIM; ++j) {
      const float u = rmax[j];
      rank += (u > v || (u == v && j > l)) ? 1 : 0;
    }
    mask[b * H_DIM + l] = (rank < RH) ? 0.0f : 1.0f;  // top-RH rows zeroed
  }
}

// Kernel 3: out = x * mask[b,h], 128-bit loads, non-temporal 128-bit stores
// (output is never re-read; keep x resident in L2 instead).
__global__ __launch_bounds__(256)
void bdt_apply(const float* __restrict__ x, const float* __restrict__ mask,
               float* __restrict__ out, int n4) {
  const int i = blockIdx.x * 256 + threadIdx.x;
  if (i >= n4) return;
  v4f v = ((const v4f*)x)[i];
  const int row = i >> 1;                 // global row index: ((b*C)+c)*H + h
  const int h = row % H_DIM;
  const int b = row / (C_DIM * H_DIM);
  const float mk = mask[b * H_DIM + h];
  v *= mk;
  __builtin_nontemporal_store(v, ((v4f*)out) + i);
}

extern "C" void kernel_launch(void* const* d_in, const int* in_sizes, int n_in,
                              void* d_out, int out_size, void* d_ws, size_t ws_size,
                              hipStream_t stream) {
  const float* x   = (const float*)d_in[0];
  float*       out = (float*)d_out;

  // workspace layout: psum[64*8*192] then mask[64*24]  (~400 KB total)
  float* psum = (float*)d_ws;
  float* mask = psum + (size_t)B_DIM * NCHUNK * HW;

  dim3 g1(B_DIM, NTILE, NCHUNK);
  bdt_stats_wmma<<<g1, 32, 0, stream>>>(x, psum);
  bdt_topk_mask<<<B_DIM, 32, 0, stream>>>(psum, mask);

  const int n4 = in_sizes[0] / 4;   // 6,291,456 float4s
  bdt_apply<<<(n4 + 255) / 256, 256, 0, stream>>>(x, mask, out, n4);
}
